// RecurrentRGCN_59339268162230
// MI455X (gfx1250) — matrix operand order repaired and checked by reading, not compiled, not run on hardware
//
#include <hip/hip_runtime.h>
#include <hip/hip_bf16.h>
#include <math.h>

// ---------------- problem constants (from reference) ----------------
#define NUM_ENTS   100000
#define NUM_RELS2  960
#define HDIM       128
#define TSTEPS     4
#define NEDGE      300000      // 18750 * 16
#define R2E_N      400000
#define GRU_OUT    384         // 3*H
#define GRU_KX     256         // 2*H
#define SLOPE_F    0.22916666666666666f   // (1/8 + 1/3)/2

typedef __attribute__((ext_vector_type(16))) _Float16 v16h;
typedef __attribute__((ext_vector_type(8)))  _Float16 v8h;
typedef __attribute__((ext_vector_type(8)))  float    v8f;

// ---------------- WMMA fragment helpers (gfx1250, wave32) ----------------
// A fragment (f16 16x32): lane&15 = M row; lane>>4 = K interleave half.
//   halves 0..7  = K  k0 + h*8 .. +7
//   halves 8..15 = K  k0 + 16 + h*8 .. +7
__device__ __forceinline__ v16h frag_a(const _Float16* rowp, int k0, int hlf) {
    v8h lo = *(const v8h*)(rowp + k0 + hlf * 8);
    v8h hi = *(const v8h*)(rowp + k0 + 16 + hlf * 8);
    v16h a;
#pragma unroll
    for (int i = 0; i < 8; ++i) { a[i] = lo[i]; a[8 + i] = hi[i]; }
    return a;
}
// B fragment (f16 32x16): lane&15 = N; same K interleave as A:
// halves 0..7 = K k0 + h*8 .. +7, halves 8..15 = K k0 + 16 + h*8 .. +7.
// Source is transposed weight Wt[N][K].
__device__ __forceinline__ v16h frag_b(const _Float16* wt_rowp, int k0, int hlf) {
    v8h lo = *(const v8h*)(wt_rowp + k0 + hlf * 8);
    v8h hi = *(const v8h*)(wt_rowp + k0 + 16 + hlf * 8);
    v16h b;
#pragma unroll
    for (int i = 0; i < 8; ++i) { b[i] = lo[i]; b[8 + i] = hi[i]; }
    return b;
}
__device__ __forceinline__ v8f wmma_f16(v16h a, v16h b, v8f c) {
    return __builtin_amdgcn_wmma_f32_16x16x32_f16(false, a, false, b, (short)0, c,
                                                  false, false);
}

// ---------------- utility kernels ----------------
__global__ void k_zero(float* __restrict__ p, long n) {
    long i  = (long)blockIdx.x * blockDim.x + threadIdx.x;
    long st = (long)gridDim.x * blockDim.x;
    for (; i < n; i += st) p[i] = 0.0f;
}

__global__ void k_copy_f32(const float* __restrict__ in, float* __restrict__ out, long n) {
    long i  = (long)blockIdx.x * blockDim.x + threadIdx.x;
    long st = (long)gridDim.x * blockDim.x;
    for (; i < n; i += st) out[i] = in[i];
}

__global__ void k_f32_to_f16(const float* __restrict__ in, _Float16* __restrict__ out, long n) {
    long i  = (long)blockIdx.x * blockDim.x + threadIdx.x;
    long st = (long)gridDim.x * blockDim.x;
    for (; i < n; i += st) out[i] = (_Float16)in[i];
}

// Wt[n*K + k] = W[k*N + n]  (f32 -> f16 transpose, for x @ W GEMMs)
__global__ void k_transpose_f16(const float* __restrict__ W, _Float16* __restrict__ Wt,
                                int K, int N) {
    int idx = blockIdx.x * blockDim.x + threadIdx.x;
    if (idx >= K * N) return;
    int n = idx / K, k = idx % K;
    Wt[idx] = (_Float16)W[k * N + n];
}

// h = l2norm(rows of in); also emit f16 copy.  block = 128 threads (= HDIM)
__global__ void k_l2norm_rows(const float* __restrict__ in, float* __restrict__ out,
                              _Float16* __restrict__ out16) {
    int  r = blockIdx.x, j = threadIdx.x;
    float v = in[(long)r * HDIM + j];
    float s = v * v;
#pragma unroll
    for (int o = 16; o > 0; o >>= 1) s += __shfl_down(s, o, 32);
    __shared__ float red[4];
    int wid = j >> 5, lane = j & 31;
    if (lane == 0) red[wid] = s;
    __syncthreads();
    if (j == 0) red[0] = 1.0f / fmaxf(sqrtf(red[0] + red[1] + red[2] + red[3]), 1e-12f);
    __syncthreads();
    float o = v * red[0];
    out  [(long)r * HDIM + j] = o;
    out16[(long)r * HDIM + j] = (_Float16)o;
}

__global__ void k_row_invnorm(const float* __restrict__ x, float* __restrict__ invn) {
    int  r = blockIdx.x, j = threadIdx.x;
    float v = x[(long)r * HDIM + j];
    float s = v * v;
#pragma unroll
    for (int o = 16; o > 0; o >>= 1) s += __shfl_down(s, o, 32);
    __shared__ float red[4];
    int wid = j >> 5, lane = j & 31;
    if (lane == 0) red[wid] = s;
    __syncthreads();
    if (j == 0) invn[r] = 1.0f / fmaxf(sqrtf(red[0] + red[1] + red[2] + red[3]), 1e-12f);
}

// ---------------- per-relation segment sums (ragged mean) ----------------
__global__ void k_seg_accum(const int* __restrict__ r_to_e, const int* __restrict__ r_seg,
                            const float* __restrict__ h, float* __restrict__ sums,
                            float* __restrict__ cnts) {
    long g = (long)blockIdx.x * blockDim.x + threadIdx.x;
    if (g >= (long)R2E_N * 32) return;
    int entry = (int)(g >> 5), sub = (int)(g & 31);
    int e = r_to_e[entry], s = r_seg[entry];
    float4 v = *(const float4*)(h + (long)e * HDIM + sub * 4);
    float* dp = sums + (long)s * HDIM + sub * 4;
    atomicAdd(dp + 0, v.x); atomicAdd(dp + 1, v.y);
    atomicAdd(dp + 2, v.z); atomicAdd(dp + 3, v.w);
    if (sub == 0) atomicAdd(&cnts[s], 1.0f);
}

// x_input = concat(emb_rel, mean) in f16  [960, 256]
__global__ void k_build_xin(const _Float16* __restrict__ emb_rel16,
                            const float* __restrict__ sums, const float* __restrict__ cnts,
                            _Float16* __restrict__ xin) {
    int idx = blockIdx.x * blockDim.x + threadIdx.x;
    if (idx >= NUM_RELS2 * GRU_KX) return;
    int r = idx / GRU_KX, c = idx % GRU_KX;
    _Float16 v;
    if (c < HDIM) v = emb_rel16[r * HDIM + c];
    else {
        float cnt = cnts[r];
        float m = (cnt > 0.0f) ? sums[r * HDIM + (c - HDIM)] / fmaxf(cnt, 1.0f) : 0.0f;
        v = (_Float16)m;
    }
    xin[idx] = v;
}

// ---------------- generic dense GEMM tile: out[M x 128] = A16 @ Wt  ----------
// grid.x = rows/16; block = 256 (8 waves, one 16x16 N-tile each); K = HDIM.
__global__ void k_matmul(const _Float16* __restrict__ A, const _Float16* __restrict__ Wt,
                         float* __restrict__ out) {
    int rt = blockIdx.x * 16;
    int wave = threadIdx.x >> 5, lane = threadIdx.x & 31, hlf = lane >> 4, l15 = lane & 15;
    const _Float16* arow = A  + (long)(rt + l15) * HDIM;
    const _Float16* brow = Wt + (long)(wave * 16 + l15) * HDIM;
    v8f c = {};
#pragma unroll
    for (int k0 = 0; k0 < HDIM; k0 += 32)
        c = wmma_f16(frag_a(arow, k0, hlf), frag_b(brow, k0, hlf), c);
    int col = wave * 16 + l15;
#pragma unroll
    for (int v = 0; v < 8; ++v)
        out[(long)(rt + v + hlf * 8) * HDIM + col] = c[v];
}

// ---------------- edge scatter: agg[dst] += HW[src] + RW[etype] ----------------
// one 32-lane group per edge; lane handles 4 contiguous columns.
__global__ void k_edge_scatter(const int* __restrict__ src, const int* __restrict__ etype,
                               const int* __restrict__ dst, const float* __restrict__ HW,
                               const float* __restrict__ RW, float* __restrict__ agg) {
    long g = (long)blockIdx.x * blockDim.x + threadIdx.x;
    if (g >= (long)NEDGE * 32) return;
    int e = (int)(g >> 5), sub = (int)(g & 31);
    int s = src[e], et = etype[e], d = dst[e];
    float4 a = *(const float4*)(HW + (long)s  * HDIM + sub * 4);
    float4 b = *(const float4*)(RW + (long)et * HDIM + sub * 4);
    float* dp = agg + (long)d * HDIM + sub * 4;
    atomicAdd(dp + 0, a.x + b.x); atomicAdd(dp + 1, a.y + b.y);
    atomicAdd(dp + 2, a.z + b.z); atomicAdd(dp + 3, a.w + b.w);
}

// ---------------- GRU gate GEMMs (fully unrolled compile-time K) ----------------
__global__ void k_gru_gemm(const _Float16* __restrict__ xin, const _Float16* __restrict__ h016,
                           const _Float16* __restrict__ wih16, const _Float16* __restrict__ whh16,
                           const float* __restrict__ b_ih, const float* __restrict__ b_hh,
                           float* __restrict__ gx, float* __restrict__ gh) {
    int mt = blockIdx.x;
    int wave = threadIdx.x >> 5, lane = threadIdx.x & 31;
    int hlf = lane >> 4, l15 = lane & 15;
    // gx = xin @ w_ih^T + b_ih   (K = 256)
    for (int nt = wave; nt < 24; nt += 8) {
        const _Float16* arow = xin   + (long)(mt * 16 + l15) * GRU_KX;
        const _Float16* brow = wih16 + (long)(nt * 16 + l15) * GRU_KX;
        v8f c = {};
#pragma unroll
        for (int k0 = 0; k0 < GRU_KX; k0 += 32)
            c = wmma_f16(frag_a(arow, k0, hlf), frag_b(brow, k0, hlf), c);
        int col = nt * 16 + l15;
        float bv = b_ih[col];
#pragma unroll
        for (int v = 0; v < 8; ++v)
            gx[(long)(mt * 16 + v + hlf * 8) * GRU_OUT + col] = c[v] + bv;
    }
    // gh = h0 @ w_hh^T + b_hh    (K = 128)
    for (int nt = wave; nt < 24; nt += 8) {
        const _Float16* arow = h016  + (long)(mt * 16 + l15) * HDIM;
        const _Float16* brow = whh16 + (long)(nt * 16 + l15) * HDIM;
        v8f c = {};
#pragma unroll
        for (int k0 = 0; k0 < HDIM; k0 += 32)
            c = wmma_f16(frag_a(arow, k0, hlf), frag_b(brow, k0, hlf), c);
        int col = nt * 16 + l15;
        float bv = b_hh[col];
#pragma unroll
        for (int v = 0; v < 8; ++v)
            gh[(long)(mt * 16 + v + hlf * 8) * GRU_OUT + col] = c[v] + bv;
    }
}

// GRU gate math + row l2norm; h0 updated in place (one block per row).
__global__ void k_gru_gates(const float* __restrict__ gx, const float* __restrict__ gh,
                            float* __restrict__ h0, _Float16* __restrict__ h016) {
    int i = blockIdx.x, j = threadIdx.x;
    const float* px = gx + (long)i * GRU_OUT;
    const float* ph = gh + (long)i * GRU_OUT;
    float r = 1.0f / (1.0f + __expf(-(px[j]       + ph[j])));
    float z = 1.0f / (1.0f + __expf(-(px[128 + j] + ph[128 + j])));
    float n = tanhf(px[256 + j] + r * ph[256 + j]);
    float hv = h0[(long)i * HDIM + j];
    float hn = (1.0f - z) * n + z * hv;
    float s = hn * hn;
#pragma unroll
    for (int o = 16; o > 0; o >>= 1) s += __shfl_down(s, o, 32);
    __shared__ float red[4];
    int wid = j >> 5, lane = j & 31;
    if (lane == 0) red[wid] = s;
    __syncthreads();
    if (j == 0) red[0] = 1.0f / fmaxf(sqrtf(red[0] + red[1] + red[2] + red[3]), 1e-12f);
    __syncthreads();
    float o = hn * red[0];
    h0  [(long)i * HDIM + j] = o;
    h016[(long)i * HDIM + j] = (_Float16)o;
}

__global__ void k_indeg(const int* __restrict__ dst, float* __restrict__ indeg) {
    int e = blockIdx.x * blockDim.x + threadIdx.x;
    if (e < NEDGE) atomicAdd(&indeg[dst[e]], 1.0f);
}

// ---------------- self-loop GEMMs + combine: leaky_relu(agg*norm + cur@(loop|evolve))
__global__ void k_combine(const _Float16* __restrict__ in16, const _Float16* __restrict__ loopT,
                          const _Float16* __restrict__ evoT, const float* __restrict__ agg,
                          const float* __restrict__ indeg, float* __restrict__ outf,
                          _Float16* __restrict__ out16) {
    int rt = blockIdx.x * 16;
    int wave = threadIdx.x >> 5, lane = threadIdx.x & 31, hlf = lane >> 4, l15 = lane & 15;
    const _Float16* arow  = in16  + (long)(rt + l15) * HDIM;
    const _Float16* blrow = loopT + (long)(wave * 16 + l15) * HDIM;
    const _Float16* berow = evoT  + (long)(wave * 16 + l15) * HDIM;
    v8f cl = {}, ce = {};
#pragma unroll
    for (int k0 = 0; k0 < HDIM; k0 += 32) {
        v16h a = frag_a(arow, k0, hlf);
        cl = wmma_f16(a, frag_b(blrow, k0, hlf), cl);
        ce = wmma_f16(a, frag_b(berow, k0, hlf), ce);
    }
    int col = wave * 16 + l15;
#pragma unroll
    for (int v = 0; v < 8; ++v) {
        int r = rt + v + hlf * 8;
        float ind  = indeg[r];
        float loop = (ind > 0.0f) ? cl[v] : ce[v];
        float nrm  = (ind > 0.0f) ? (1.0f / fmaxf(ind, 1.0f)) : 0.0f;
        float val  = agg[(long)r * HDIM + col] * nrm + loop;
        val = (val > 0.0f) ? val : val * SLOPE_F;
        outf [(long)r * HDIM + col] = val;
        out16[(long)r * HDIM + col] = (_Float16)val;
    }
}

// ---------------- time gate: tw = sigmoid(h @ Wt + b); h' = tw*cur_norm + (1-tw)*h
__global__ void k_timegate(const _Float16* __restrict__ h16, const _Float16* __restrict__ TWt,
                           const float* __restrict__ tgb, const float* __restrict__ hin,
                           const float* __restrict__ cur, const float* __restrict__ invn,
                           float* __restrict__ hout, _Float16* __restrict__ h16out) {
    int rt = blockIdx.x * 16;
    int wave = threadIdx.x >> 5, lane = threadIdx.x & 31, hlf = lane >> 4, l15 = lane & 15;
    const _Float16* arow = h16 + (long)(rt + l15) * HDIM;
    const _Float16* brow = TWt + (long)(wave * 16 + l15) * HDIM;
    v8f c = {};
#pragma unroll
    for (int k0 = 0; k0 < HDIM; k0 += 32)
        c = wmma_f16(frag_a(arow, k0, hlf), frag_b(brow, k0, hlf), c);
    int col = wave * 16 + l15;
#pragma unroll
    for (int v = 0; v < 8; ++v) {
        int r = rt + v + hlf * 8;
        float tw = 1.0f / (1.0f + __expf(-(c[v] + tgb[col])));
        float ch = cur[(long)r * HDIM + col] * invn[r];
        float hv = hin[(long)r * HDIM + col];
        float o  = tw * ch + (1.0f - tw) * hv;
        hout  [(long)r * HDIM + col] = o;
        h16out[(long)r * HDIM + col] = (_Float16)o;
    }
}

// ---------------- host side ----------------
extern "C" void kernel_launch(void* const* d_in, const int* in_sizes, int n_in,
                              void* d_out, int out_size, void* d_ws, size_t ws_size,
                              hipStream_t stream) {
    const int*   src        = (const int*)  d_in[0];
    const int*   dst        = (const int*)  d_in[1];
    const int*   etype      = (const int*)  d_in[2];
    const int*   r_to_e     = (const int*)  d_in[3];
    const int*   r_seg      = (const int*)  d_in[4];
    const float* dyn_emb    = (const float*)d_in[5];
    const float* emb_rel    = (const float*)d_in[6];
    const float* tg_w       = (const float*)d_in[7];
    const float* tg_b       = (const float*)d_in[8];
    const float* gru_w_ih   = (const float*)d_in[9];
    const float* gru_w_hh   = (const float*)d_in[10];
    const float* gru_b_ih   = (const float*)d_in[11];
    const float* gru_b_hh   = (const float*)d_in[12];
    const float* w_neigh    = (const float*)d_in[13];
    const float* w_loop     = (const float*)d_in[14];
    const float* w_evo      = (const float*)d_in[15];
    float* out = (float*)d_out;

    // bump allocator over d_ws (~237 MB total)
    char* wp = (char*)d_ws;
    auto alloc = [&](size_t bytes) -> void* {
        void* p = (void*)wp;
        wp += (bytes + 255) & ~(size_t)255;
        return p;
    };
    float*    hbuf0  = (float*)   alloc((size_t)NUM_ENTS * HDIM * 4);
    _Float16* hf16A  = (_Float16*)alloc((size_t)NUM_ENTS * HDIM * 2);
    _Float16* hf16B  = (_Float16*)alloc((size_t)NUM_ENTS * HDIM * 2);
    _Float16* cur16A = (_Float16*)alloc((size_t)NUM_ENTS * HDIM * 2);
    float*    curbuf = (float*)   alloc((size_t)NUM_ENTS * HDIM * 4);  // doubles as HW
    float*    agg    = (float*)   alloc((size_t)NUM_ENTS * HDIM * 4);
    float*    indeg  = (float*)   alloc((size_t)NUM_ENTS * 4);
    float*    invn   = (float*)   alloc((size_t)NUM_ENTS * 4);
    float*    RW     = (float*)   alloc((size_t)NUM_RELS2 * HDIM * 4);
    float*    sums   = (float*)   alloc((size_t)NUM_RELS2 * HDIM * 4);
    float*    cnts   = (float*)   alloc((size_t)NUM_RELS2 * 4);
    float*    h0buf  = (float*)   alloc((size_t)NUM_RELS2 * HDIM * 4);
    _Float16* h016   = (_Float16*)alloc((size_t)NUM_RELS2 * HDIM * 2);
    _Float16* embr16 = (_Float16*)alloc((size_t)NUM_RELS2 * HDIM * 2);
    _Float16* xin16  = (_Float16*)alloc((size_t)NUM_RELS2 * GRU_KX * 2);
    float*    gx     = (float*)   alloc((size_t)NUM_RELS2 * GRU_OUT * 4);
    float*    gh     = (float*)   alloc((size_t)NUM_RELS2 * GRU_OUT * 4);
    _Float16* WnT    = (_Float16*)alloc((size_t)2 * HDIM * HDIM * 2);
    _Float16* loopT  = (_Float16*)alloc((size_t)2 * HDIM * HDIM * 2);
    _Float16* evoT   = (_Float16*)alloc((size_t)2 * HDIM * HDIM * 2);
    _Float16* TWt    = (_Float16*)alloc((size_t)HDIM * HDIM * 2);
    _Float16* wih16  = (_Float16*)alloc((size_t)GRU_OUT * GRU_KX * 2);
    _Float16* whh16  = (_Float16*)alloc((size_t)GRU_OUT * HDIM * 2);

    const int ZB = 2048;     // grid-stride zero blocks
    const int HH = HDIM * HDIM;

    // ---- one-time (per launch) precompute ----
    k_l2norm_rows<<<NUM_ENTS, 128, 0, stream>>>(dyn_emb, hbuf0, hf16A);
    for (int l = 0; l < 2; ++l) {
        k_transpose_f16<<<(HH + 255) / 256, 256, 0, stream>>>(w_neigh + l * HH, WnT   + l * HH, HDIM, HDIM);
        k_transpose_f16<<<(HH + 255) / 256, 256, 0, stream>>>(w_loop  + l * HH, loopT + l * HH, HDIM, HDIM);
        k_transpose_f16<<<(HH + 255) / 256, 256, 0, stream>>>(w_evo   + l * HH, evoT  + l * HH, HDIM, HDIM);
    }
    k_transpose_f16<<<(HH + 255) / 256, 256, 0, stream>>>(tg_w, TWt, HDIM, HDIM);
    k_f32_to_f16<<<192, 256, 0, stream>>>(gru_w_ih, wih16, (long)GRU_OUT * GRU_KX);
    k_f32_to_f16<<<192, 256, 0, stream>>>(gru_w_hh, whh16, (long)GRU_OUT * HDIM);
    k_f32_to_f16<<<192, 256, 0, stream>>>(emb_rel, embr16, (long)NUM_RELS2 * HDIM);
    k_f32_to_f16<<<192, 256, 0, stream>>>(emb_rel, h016,   (long)NUM_RELS2 * HDIM);
    k_copy_f32  <<<192, 256, 0, stream>>>(emb_rel, h0buf,  (long)NUM_RELS2 * HDIM);

    const long NE_H = (long)NUM_ENTS * HDIM;
    for (int t = 0; t < TSTEPS; ++t) {
        const int*   src_t  = src    + (long)t * NEDGE;
        const int*   dst_t  = dst    + (long)t * NEDGE;
        const int*   ety_t  = etype  + (long)t * NEDGE;
        const int*   rte_t  = r_to_e + (long)t * R2E_N;
        const int*   rsg_t  = r_seg  + (long)t * R2E_N;
        const float* hin    = (t == 0) ? hbuf0 : (out + (long)(t - 1) * NE_H);
        float*       hout   = out + (long)t * NE_H;
        _Float16*    h16in  = (t & 1) ? hf16B : hf16A;
        _Float16*    h16out = (t & 1) ? hf16A : hf16B;

        // ---- ragged per-relation mean -> GRU relation update ----
        k_zero<<<ZB, 256, 0, stream>>>(sums, (long)NUM_RELS2 * HDIM);
        k_zero<<<4, 256, 0, stream>>>(cnts, NUM_RELS2);
        k_zero<<<ZB, 256, 0, stream>>>(indeg, NUM_ENTS);
        k_seg_accum<<<(R2E_N * 32) / 256, 256, 0, stream>>>(rte_t, rsg_t, hin, sums, cnts);
        k_build_xin<<<(NUM_RELS2 * GRU_KX + 255) / 256, 256, 0, stream>>>(embr16, sums, cnts, xin16);
        k_gru_gemm<<<NUM_RELS2 / 16, 256, 0, stream>>>(xin16, h016, wih16, whh16,
                                                       gru_b_ih, gru_b_hh, gx, gh);
        k_gru_gates<<<NUM_RELS2, 128, 0, stream>>>(gx, gh, h0buf, h016);

        k_indeg<<<(NEDGE + 255) / 256, 256, 0, stream>>>(dst_t, indeg);

        // ---- RGCN layer 1:  (h+r)@W  ==  (h@W)[src] + (r@W)[etype] ----
        k_matmul<<<NUM_ENTS / 16, 256, 0, stream>>>(h16in, WnT, curbuf);   // HW
        k_matmul<<<NUM_RELS2 / 16, 256, 0, stream>>>(h016, WnT, RW);       // RW
        k_zero<<<ZB, 256, 0, stream>>>(agg, NE_H);
        k_edge_scatter<<<(NEDGE * 32) / 256, 256, 0, stream>>>(src_t, ety_t, dst_t,
                                                               curbuf, RW, agg);
        k_combine<<<NUM_ENTS / 16, 256, 0, stream>>>(h16in, loopT, evoT, agg, indeg,
                                                     curbuf, cur16A);
        // ---- RGCN layer 2 ----
        k_matmul<<<NUM_ENTS / 16, 256, 0, stream>>>(cur16A, WnT + HH, curbuf);
        k_matmul<<<NUM_RELS2 / 16, 256, 0, stream>>>(h016, WnT + HH, RW);
        k_zero<<<ZB, 256, 0, stream>>>(agg, NE_H);
        k_edge_scatter<<<(NEDGE * 32) / 256, 256, 0, stream>>>(src_t, ety_t, dst_t,
                                                               curbuf, RW, agg);
        // h16out is fully overwritten by k_timegate below -> safe f16 sink here
        k_combine<<<NUM_ENTS / 16, 256, 0, stream>>>(cur16A, loopT + HH, evoT + HH,
                                                     agg, indeg, curbuf, h16out);

        // ---- l2norm(cur) + time gate -> h_t ----
        k_row_invnorm<<<NUM_ENTS, 128, 0, stream>>>(curbuf, invn);
        k_timegate<<<NUM_ENTS / 16, 256, 0, stream>>>(h16in, TWt, tg_b, hin, curbuf,
                                                      invn, hout, h16out);
    }
    (void)in_sizes; (void)n_in; (void)out_size; (void)ws_size;
}
